// LlamaDecoderLayer_2_91096256348257
// MI455X (gfx1250) — matrix operand order repaired and checked by reading, not compiled, not run on hardware
//
#include <hip/hip_runtime.h>
#include <math.h>

#define Bn   4
#define Sn   2048
#define Hn   4096
#define En   64
#define Kc   4
#define Tn   4
#define Dn   100
#define KIn  2048
#define KTOT 4224      // H + 128 (D padded to 128), multiple of 32
#define DPAD 128
#define ASTR 48        // LDS row stride (bf16): 96B, multiple of 16B for b128 alignment
#define BSTR 48

typedef __attribute__((ext_vector_type(16))) __bf16 v16bf;
typedef __attribute__((ext_vector_type(8)))  float  v8f;

// ---- optional CDNA5 async global->LDS staging (guarded; falls back to sync copies)
#if defined(__AMDGCN__) && __has_builtin(__builtin_amdgcn_global_load_async_to_lds_b128) && \
    __has_builtin(__builtin_amdgcn_s_wait_asynccnt)
#define USE_ASYNC_LDS 1
typedef int v4i __attribute__((__vector_size__(4 * sizeof(int))));
typedef __attribute__((address_space(1))) v4i* g_b128p;   // AS(1) prints as __device__
typedef __attribute__((address_space(3))) v4i* l_b128p;   // LDS
#else
#define USE_ASYNC_LDS 0
#endif

// copy 8 bf16 (16B) global -> LDS
__device__ __forceinline__ void cp16(const __bf16* g, __bf16* l) {
#if USE_ASYNC_LDS
  __builtin_amdgcn_global_load_async_to_lds_b128((g_b128p)g, (l_b128p)l, 0, 0);
#else
#pragma unroll
  for (int i = 0; i < 8; ++i) l[i] = g[i];
#endif
}
__device__ __forceinline__ void async_fence() {
#if USE_ASYNC_LDS
  __builtin_amdgcn_s_wait_asynccnt(0);
#endif
}

__device__ __forceinline__ __bf16 f2bf(float f) {
  unsigned u = __float_as_uint(f);
  unsigned r = (u + 0x7FFFu + ((u >> 16) & 1u)) >> 16;   // round-nearest-even
  return __builtin_bit_cast(__bf16, (unsigned short)r);
}
__device__ __forceinline__ float bf2f(__bf16 x) {
  unsigned short us = __builtin_bit_cast(unsigned short, x);
  return __uint_as_float(((unsigned)us) << 16);
}
__device__ __forceinline__ float silu(float x) { return x / (1.f + __expf(-x)); }

// ---------------- RMSNorm: hidden [B,S,H] -> hs_bf [B,S+1,H] bf16, pad row = 0
__global__ void __launch_bounds__(256) k_rmsnorm(const float* __restrict__ x,
                                                 const float* __restrict__ w,
                                                 __bf16* __restrict__ hs_bf) {
  int bs = blockIdx.x;                 // over B*(S+1)
  int b = bs / (Sn + 1), s = bs % (Sn + 1);
  int tid = threadIdx.x;
  __bf16* orow = hs_bf + (size_t)bs * Hn;
  if (s == Sn) {
    __bf16 z = f2bf(0.f);
    for (int h = tid; h < Hn; h += 256) orow[h] = z;
    return;
  }
  const float* xr = x + ((size_t)b * Sn + s) * Hn;
  float ss = 0.f;
  for (int h = tid; h < Hn; h += 256) { float v = xr[h]; ss += v * v; }
  __shared__ float red[256];
  red[tid] = ss; __syncthreads();
  for (int sft = 128; sft > 0; sft >>= 1) {
    if (tid < sft) red[tid] += red[tid + sft];
    __syncthreads();
  }
  float rs = rsqrtf(red[0] / (float)Hn + 1e-6f);
  for (int h = tid; h < Hn; h += 256) orow[h] = f2bf(w[h] * xr[h] * rs);
}

// ---------------- W transpose+pad: mlp_w [4196,4096] f32 -> Wt [4096,KTOT] bf16
__global__ void __launch_bounds__(256) k_transpose_w(const float* __restrict__ mlp_w,
                                                     __bf16* __restrict__ Wt) {
  __shared__ float tile[32][33];
  int k0 = blockIdx.x * 32, n0 = blockIdx.y * 32;
  int tx = threadIdx.x & 31, ty = threadIdx.x >> 5;   // ty = 0..7
#pragma unroll
  for (int r = 0; r < 4; ++r) {
    int k = k0 + ty + r * 8;
    tile[ty + r * 8][tx] = (k < Hn + Dn) ? mlp_w[(size_t)k * Hn + (n0 + tx)] : 0.f;
  }
  __syncthreads();
#pragma unroll
  for (int r = 0; r < 4; ++r) {
    int n = n0 + ty + r * 8;
    Wt[(size_t)n * KTOT + (k0 + tx)] = f2bf(tile[tx][ty + r * 8]);
  }
}

__global__ void __launch_bounds__(256) k_cvt_down(const float* __restrict__ dw,
                                                  __bf16* __restrict__ d_bf) {
  size_t i = ((size_t)blockIdx.x * 256 + threadIdx.x) * 4;   // total KIn*Hn, exact
#pragma unroll
  for (int j = 0; j < 4; ++j) d_bf[i + j] = f2bf(dw[i + j]);
}
__global__ void __launch_bounds__(256) k_zero_tmp(__bf16* __restrict__ tmp) {
  size_t i = (size_t)blockIdx.x * 256 + threadIdx.x;         // total B*(S+1)*128, exact
  tmp[i] = f2bf(0.f);
}

// ---------------- per-entity subtoken mean: b_vec [256, H] bf16
__global__ void __launch_bounds__(256) k_bvec(const int* __restrict__ wst,
                                              const __bf16* __restrict__ hs_bf,
                                              __bf16* __restrict__ bvec) {
  int eg = blockIdx.x;                  // b*64 + e
  int b = eg >> 6, e = eg & 63;
  int tid = threadIdx.x;
  int pos[Tn]; int cnt = 0;
#pragma unroll
  for (int t = 0; t < Tn; ++t) {
    int p = wst[((size_t)b * En + e) * Tn + t];
    if (p >= 0) cnt++;
    pos[t] = (p < 0) ? Sn : p;          // pad row is zero
  }
  float inv = 1.f / (float)((cnt > 0) ? cnt : 1);
  const __bf16* base = hs_bf + (size_t)b * (Sn + 1) * Hn;
  for (int h = tid; h < Hn; h += 256) {
    float s = 0.f;
#pragma unroll
    for (int t = 0; t < Tn; ++t) s += bf2f(base[(size_t)pos[t] * Hn + h]);
    bvec[(size_t)eg * Hn + h] = f2bf(s * inv);
  }
}

// ---------------- WMMA fragment helpers (ISA 7.12.2 layouts)
__device__ __forceinline__ v16bf load_a_frag(const __bf16* As, int m_off, int lane) {
  int r  = lane & 15;
  int kb = (lane < 16) ? 0 : 8;
  const __bf16* row = As + (size_t)(m_off + r) * ASTR;
  v16bf a;
#pragma unroll
  for (int i = 0; i < 8; ++i) { a[i] = row[kb + i]; a[i + 8] = row[kb + 16 + i]; }
  return a;
}
__device__ __forceinline__ v16bf load_b_frag(const __bf16* Bt, int n_off, int lane) {
  const __bf16* col = Bt + (size_t)(n_off + (lane & 15)) * BSTR + ((lane < 16) ? 0 : 8);
  v16bf bf;
#pragma unroll
  for (int i = 0; i < 8; ++i) { bf[i] = col[i]; bf[i + 8] = col[16 + i]; }
  return bf;
}

// ---------------- db GEMM: db[256,KI] = bvec[256,H] @ down_w^T   (B(k,n)=down[n,k])
__global__ void __launch_bounds__(256) k_db(const __bf16* __restrict__ A,
                                            const __bf16* __restrict__ Dw,
                                            float* __restrict__ db) {
  __shared__ __bf16 As[64 * ASTR];
  __shared__ __bf16 Bt[128 * BSTR];
  int tid = threadIdx.x, lane = tid & 31, wv = tid >> 5;
  int wm = (wv >> 2) * 32, wn = (wv & 3) * 32;
  int bm = blockIdx.y * 64, bn = blockIdx.x * 128;
  v8f z = {0.f,0.f,0.f,0.f,0.f,0.f,0.f,0.f};
  v8f acc00 = z, acc01 = z, acc10 = z, acc11 = z;
  int sa_row = tid >> 2, sa_k = (tid & 3) * 8;
  int sb_n = tid >> 1, sb_k = (tid & 1) * 16;
  for (int kt = 0; kt < Hn / 32; ++kt) {
    int k0 = kt * 32;
    async_fence();
    __syncthreads();
    cp16(A + (size_t)(bm + sa_row) * Hn + k0 + sa_k, As + sa_row * ASTR + sa_k);
    cp16(Dw + (size_t)(bn + sb_n) * Hn + k0 + sb_k,     Bt + sb_n * BSTR + sb_k);
    cp16(Dw + (size_t)(bn + sb_n) * Hn + k0 + sb_k + 8, Bt + sb_n * BSTR + sb_k + 8);
    async_fence();
    __syncthreads();
    v16bf a0 = load_a_frag(As, wm, lane);
    v16bf a1 = load_a_frag(As, wm + 16, lane);
    v16bf b0 = load_b_frag(Bt, wn, lane);
    v16bf b1 = load_b_frag(Bt, wn + 16, lane);
    acc00 = __builtin_amdgcn_wmma_f32_16x16x32_bf16(false, a0, false, b0, (short)0, acc00, false, false);
    acc01 = __builtin_amdgcn_wmma_f32_16x16x32_bf16(false, a0, false, b1, (short)0, acc01, false, false);
    acc10 = __builtin_amdgcn_wmma_f32_16x16x32_bf16(false, a1, false, b0, (short)0, acc10, false, false);
    acc11 = __builtin_amdgcn_wmma_f32_16x16x32_bf16(false, a1, false, b1, (short)0, acc11, false, false);
  }
  v8f accs[2][2] = {{acc00, acc01}, {acc10, acc11}};
#pragma unroll
  for (int mi = 0; mi < 2; ++mi)
#pragma unroll
    for (int ni = 0; ni < 2; ++ni) {
      v8f c = accs[mi][ni];
#pragma unroll
      for (int r = 0; r < 8; ++r) {
        int m = bm + wm + mi * 16 + r + 8 * (lane >> 4);
        int n = bn + wn + ni * 16 + (lane & 15);
        db[(size_t)m * KIn + n] = c[r];
      }
    }
}

// ---------------- entity kernel: gate/up/silu, logits via db, softmax, attn, scatter
__global__ void __launch_bounds__(256) k_entity(const int* __restrict__ we,
                                                const int* __restrict__ wst,
                                                const float* __restrict__ concept,
                                                const float* __restrict__ sentinel,
                                                const float* __restrict__ gate_w,
                                                const float* __restrict__ up_w,
                                                const float* __restrict__ db,
                                                __bf16* __restrict__ tmp) {
  int eg = blockIdx.x;                 // b*64 + e
  int b = eg >> 6, e = eg & 63;
  int tid = threadIdx.x;
  __shared__ float ent[Kc + 1][Dn];
  __shared__ float red[256];
  __shared__ float logits[Kc + 1];
  __shared__ float attn[Kc + 1];
  for (int i = tid; i < (Kc + 1) * Dn; i += 256) {
    int k = i / Dn, d = i % Dn;
    float v;
    if (k < Kc) {
      int c = we[((size_t)b * En + e) * Kc + k];
      c = (c < 0) ? 0 : c;
      v = concept[(size_t)c * Dn + d];
    } else v = sentinel[d];
    ent[k][d] = v;
  }
  __syncthreads();
  const float* dbr = db + (size_t)eg * KIn;
  for (int k = 0; k < Kc + 1; ++k) {
    float part = 0.f;
#pragma unroll
    for (int j = 0; j < KIn / 256; ++j) {
      int ki = tid + j * 256;          // coalesced across threads
      float g = 0.f, u = 0.f;
      for (int d = 0; d < Dn; ++d) {
        float ev = ent[k][d];
        g = fmaf(ev, gate_w[(size_t)d * KIn + ki], g);
        u = fmaf(ev, up_w[(size_t)d * KIn + ki], u);
      }
      part = fmaf(silu(g) * u, dbr[ki], part);
    }
    red[tid] = part; __syncthreads();
    for (int sft = 128; sft > 0; sft >>= 1) {
      if (tid < sft) red[tid] += red[tid + sft];
      __syncthreads();
    }
    if (tid == 0) {
      float m = 0.f;
      if (k < Kc && we[((size_t)b * En + e) * Kc + k] == -1) m = -1e9f;
      logits[k] = red[0] + m;
    }
    __syncthreads();
  }
  if (tid == 0) {
    float mx = logits[0];
#pragma unroll
    for (int k = 1; k < Kc + 1; ++k) mx = fmaxf(mx, logits[k]);
    float ssum = 0.f, ex[Kc + 1];
#pragma unroll
    for (int k = 0; k < Kc + 1; ++k) { ex[k] = __expf(logits[k] - mx); ssum += ex[k]; }
#pragma unroll
    for (int k = 0; k < Kc + 1; ++k) attn[k] = ex[k] / ssum;
  }
  __syncthreads();
  if (tid < Dn) {
    float ao = 0.f;
#pragma unroll
    for (int k = 0; k < Kc + 1; ++k) ao = fmaf(attn[k], ent[k][tid], ao);
    __bf16 aob = f2bf(ao);
#pragma unroll
    for (int t = 0; t < Tn; ++t) {
      int p = wst[((size_t)b * En + e) * Tn + t];
      int pos = (p < 0) ? Sn : p;      // pad row is dropped downstream
      tmp[((size_t)b * (Sn + 1) + pos) * DPAD + tid] = aob;
    }
  }
}

// ---------------- main MLP GEMM: out = silu([hs|tmp] @ W + bias)*alpha + resid
// 128x128 block tile, 8 wave32 waves each 32x64 (2x4 wmma), double-buffered LDS.
__global__ void __launch_bounds__(256) k_mlp(const __bf16* __restrict__ A,   // [B,S+1,H]
                                             const __bf16* __restrict__ tmp, // [B,S+1,128]
                                             const __bf16* __restrict__ Wt,  // [Hn,KTOT] (N-major)
                                             const float* __restrict__ bias,
                                             const float* __restrict__ alphaPtr,
                                             const float* __restrict__ resid,
                                             float* __restrict__ out) {
  __shared__ __bf16 As[2][128 * ASTR];
  __shared__ __bf16 Bt[2][128 * BSTR];
  int tid = threadIdx.x, lane = tid & 31, wv = tid >> 5;
  int wm = (wv >> 1) * 32;             // 0,32,64,96
  int wn = (wv & 1) * 64;              // 0,64
  int bm = blockIdx.y * 128, bn = blockIdx.x * 128;
  v8f z = {0.f,0.f,0.f,0.f,0.f,0.f,0.f,0.f};
  v8f acc[2][4];
#pragma unroll
  for (int mi = 0; mi < 2; ++mi)
#pragma unroll
    for (int ni = 0; ni < 4; ++ni) acc[mi][ni] = z;

  int srow = tid >> 1, skseg = (tid & 1) * 16;      // 128 rows x 32 k, 16 elems/thread
  int bs = bm + srow;
  int bb = bs >> 11, ssx = bs & (Sn - 1);
  const __bf16* a_hs = A + ((size_t)bb * (Sn + 1) + ssx) * Hn + skseg;
  const __bf16* a_tm = tmp + ((size_t)bb * (Sn + 1) + ssx) * DPAD + skseg;
  const __bf16* b_src = Wt + (size_t)(bn + srow) * KTOT + skseg;
  __bf16* da = &As[0][srow * ASTR + skseg];
  __bf16* dbp = &Bt[0][srow * BSTR + skseg];
  const int lbuf = 128 * ASTR;

  auto stage = [&](int kt, int buf) {
    int k0 = kt * 32;
    const __bf16* sa = (k0 < Hn) ? (a_hs + k0) : (a_tm + (k0 - Hn));
    cp16(sa,     da + buf * lbuf);
    cp16(sa + 8, da + buf * lbuf + 8);
    const __bf16* sb = b_src + k0;
    cp16(sb,     dbp + buf * lbuf);
    cp16(sb + 8, dbp + buf * lbuf + 8);
  };

  const int NT = KTOT / 32;
  stage(0, 0);
  for (int kt = 0; kt < NT; ++kt) {
    async_fence();                                    // our async stage(kt) complete
    __syncthreads();                                  // visible to all; buf (kt+1)&1 free
    if (kt + 2 < NT) __builtin_prefetch(b_src + (size_t)(kt + 2) * 32, 0, 1);
    if (kt + 1 < NT) stage(kt + 1, (kt + 1) & 1);
    int buf = kt & 1;
    v16bf af0 = load_a_frag(&As[buf][0], wm, lane);
    v16bf af1 = load_a_frag(&As[buf][0], wm + 16, lane);
    v16bf bf_[4];
#pragma unroll
    for (int ni = 0; ni < 4; ++ni) bf_[ni] = load_b_frag(&Bt[buf][0], wn + ni * 16, lane);
#pragma unroll
    for (int ni = 0; ni < 4; ++ni) {
      acc[0][ni] = __builtin_amdgcn_wmma_f32_16x16x32_bf16(false, af0, false, bf_[ni], (short)0, acc[0][ni], false, false);
      acc[1][ni] = __builtin_amdgcn_wmma_f32_16x16x32_bf16(false, af1, false, bf_[ni], (short)0, acc[1][ni], false, false);
    }
  }
  float alpha = alphaPtr[0];
#pragma unroll
  for (int mi = 0; mi < 2; ++mi)
#pragma unroll
    for (int ni = 0; ni < 4; ++ni) {
      v8f c = acc[mi][ni];
#pragma unroll
      for (int r = 0; r < 8; ++r) {
        int m = bm + wm + mi * 16 + r + 8 * (lane >> 4);   // row over B*S
        int n = bn + wn + ni * 16 + (lane & 15);
        float x = c[r] + bias[n];
        float s = silu(x);
        out[(size_t)m * Hn + n] = s * alpha + resid[(size_t)m * Hn + n];
      }
    }
}

extern "C" void kernel_launch(void* const* d_in, const int* in_sizes, int n_in,
                              void* d_out, int out_size, void* d_ws, size_t ws_size,
                              hipStream_t stream) {
  (void)in_sizes; (void)n_in; (void)out_size; (void)ws_size;
  const float* hidden   = (const float*)d_in[0];
  const int*   we       = (const int*)d_in[1];
  const int*   wst      = (const int*)d_in[2];
  const float* concept  = (const float*)d_in[3];
  const float* sentinel = (const float*)d_in[4];
  const float* ln_w     = (const float*)d_in[5];
  const float* gate_w   = (const float*)d_in[6];
  const float* up_w     = (const float*)d_in[7];
  const float* down_w   = (const float*)d_in[8];
  const float* mlp_w    = (const float*)d_in[9];
  const float* mlp_b    = (const float*)d_in[10];
  const float* alpha    = (const float*)d_in[11];
  float* out = (float*)d_out;

  char* ws = (char*)d_ws;
  size_t off = 0;
  auto take = [&](size_t bytes) { char* p = ws + off; off += (bytes + 255) & ~(size_t)255; return p; };
  __bf16* hs_bf   = (__bf16*)take((size_t)Bn * (Sn + 1) * Hn * 2);
  __bf16* wt_bf   = (__bf16*)take((size_t)Hn * KTOT * 2);
  __bf16* down_bf = (__bf16*)take((size_t)KIn * Hn * 2);
  __bf16* bvec    = (__bf16*)take((size_t)Bn * En * Hn * 2);
  float*  db      = (float*) take((size_t)Bn * En * KIn * 4);
  __bf16* tmp     = (__bf16*)take((size_t)Bn * (Sn + 1) * DPAD * 2);

  k_rmsnorm    <<<Bn * (Sn + 1), 256, 0, stream>>>(hidden, ln_w, hs_bf);
  k_transpose_w<<<dim3(KTOT / 32, Hn / 32), 256, 0, stream>>>(mlp_w, wt_bf);
  k_cvt_down   <<<(int)(((size_t)KIn * Hn) / 1024), 256, 0, stream>>>(down_w, down_bf);
  k_zero_tmp   <<<(int)(((size_t)Bn * (Sn + 1) * DPAD) / 256), 256, 0, stream>>>(tmp);
  k_bvec       <<<Bn * En, 256, 0, stream>>>(wst, hs_bf, bvec);
  k_db         <<<dim3(KIn / 128, (Bn * En) / 64), 256, 0, stream>>>(bvec, down_bf, db);
  k_entity     <<<Bn * En, 256, 0, stream>>>(we, wst, concept, sentinel, gate_w, up_w, db, tmp);
  k_mlp        <<<dim3(Hn / 128, (Bn * Sn) / 128), 256, 0, stream>>>(hs_bf, tmp, wt_bf, mlp_b, alpha, hidden, out);
}
